// SiTMAEUnpadFlashAttention_89799176225219
// MI455X (gfx1250) — matrix operand order repaired and checked by reading, not compiled, not run on hardware
//
#include <hip/hip_runtime.h>
#include <hip/hip_bf16.h>

// ---------------------------------------------------------------------------
// CDNA5 (gfx1250) bf16 WMMA implementation of unpadded MHA block:
//   qkv = X @ Wqkv^T + bq ; per-seq softmax(Q K^T / 8) V ; out = ctx @ Wo^T + bo
// Matmuls on v_wmma_f32_16x16x32_bf16; K-tile staging via async global->LDS.
// ---------------------------------------------------------------------------

typedef __attribute__((ext_vector_type(16))) __bf16         v16bf;
typedef __attribute__((ext_vector_type(8)))  float          v8f;
typedef __attribute__((ext_vector_type(16))) unsigned short v16u;
typedef __attribute__((ext_vector_type(8)))  unsigned short v8u;

__device__ __forceinline__ unsigned short f32_to_bf16_rne(float f) {
  union { float f; unsigned u; } v; v.f = f;
  unsigned r = v.u + 0x7FFFu + ((v.u >> 16) & 1u);
  return (unsigned short)(r >> 16);
}

__device__ __forceinline__ int imin(int a, int b) { return a < b ? a : b; }

// A operand (16x32 bf16, M=lane&15). Layout: lanes 0-15 hold K={0..15},
// lanes 16-31 hold K={16..31}.  row = pointer to this lane's M-row.
__device__ __forceinline__ v16bf load_a_frag(const unsigned short* row, int k0, int hi) {
  const v8u lo = *(const v8u*)(row + k0 + hi * 16);
  const v8u up = *(const v8u*)(row + k0 + hi * 16 + 8);
  v16u r;
#pragma unroll
  for (int i = 0; i < 8; ++i) { r[i] = lo[i]; r[i + 8] = up[i]; }
  return __builtin_bit_cast(v16bf, r);
}

// B operand (32x16 bf16, N=lane&15): 16 contiguous K values per lane.
__device__ __forceinline__ v16bf load_b_frag(const unsigned short* p) {
  return __builtin_bit_cast(v16bf, *(const v16u*)p);
}

__device__ __forceinline__ v8f wmma_bf16(v16bf a, v16bf b, v8f c) {
  return __builtin_amdgcn_wmma_f32_16x16x32_bf16(false, a, false, b, (short)0, c,
                                                 false, false);
}

// gfx1250 async copy: 16B global -> LDS, tracked by ASYNCcnt (no VGPR bounce).
__device__ __forceinline__ void async_copy_b128(unsigned lds_off, const void* gaddr) {
  asm volatile("global_load_async_to_lds_b128 %0, %1, off"
               :: "v"(lds_off), "v"(gaddr) : "memory");
}
__device__ __forceinline__ void wait_asynccnt0() {
  asm volatile("s_wait_asynccnt 0" ::: "memory");
}

// ---------------------------------------------------------------------------
__global__ void cvt_f32_bf16(const float* __restrict__ src,
                             unsigned short* __restrict__ dst, int n) {
  int i = blockIdx.x * blockDim.x + threadIdx.x;
  int stride = gridDim.x * blockDim.x;
  for (; i < n; i += stride) dst[i] = f32_to_bf16_rne(src[i]);
}

// ---------------------------------------------------------------------------
// C[M,N] = A[M,K](bf16) * W[N,K]^T(bf16) + bias ; M%128==0, N%256==0, K%32==0.
// Block 256 thr = 8 waves; block tile 128x256; wave tile 64x64 (4x4 frags):
// 16 WMMAs per K-step of 32 against 16x16B loads/lane -> ~32 FLOP/B intensity.
template <bool BF16_OUT>
__global__ __launch_bounds__(256) void gemm_xwT(
    const unsigned short* __restrict__ A, const unsigned short* __restrict__ W,
    const float* __restrict__ bias, void* __restrict__ Cout, int M, int N, int K) {
  const int lane  = threadIdx.x & 31;
  const int wid   = threadIdx.x >> 5;
  const int waveM = wid & 1;           // 0..1 (64 rows each)
  const int waveN = wid >> 1;          // 0..3 (64 cols each)
  const int hi    = lane >> 4;
  const int ln    = lane & 15;

  const int mbase = blockIdx.x * 128 + waveM * 64;
  const int nbase = blockIdx.y * 256 + waveN * 64;

  v8f acc[4][4];
#pragma unroll
  for (int i = 0; i < 4; ++i)
#pragma unroll
    for (int j = 0; j < 4; ++j)
#pragma unroll
      for (int r = 0; r < 8; ++r) acc[i][j][r] = 0.f;

  const unsigned short* arow[4];
  const unsigned short* wrow[4];
#pragma unroll
  for (int i = 0; i < 4; ++i) {
    arow[i] = A + (size_t)(mbase + i * 16 + ln) * K;
    wrow[i] = W + (size_t)(nbase + i * 16 + ln) * K;
  }

  for (int kk = 0; kk < K; kk += 32) {
    __builtin_prefetch(arow[0] + kk + 256, 0, 0);
    v16bf a[4];
#pragma unroll
    for (int i = 0; i < 4; ++i) a[i] = load_a_frag(arow[i], kk, hi);
#pragma unroll
    for (int f = 0; f < 4; ++f) {
      const v16bf b = load_b_frag(wrow[f] + kk + hi * 16);
#pragma unroll
      for (int i = 0; i < 4; ++i) acc[i][f] = wmma_bf16(a[i], b, acc[i][f]);
    }
  }

#pragma unroll
  for (int fm = 0; fm < 4; ++fm)
#pragma unroll
    for (int fn = 0; fn < 4; ++fn) {
      const int n  = nbase + fn * 16 + ln;
      const float bv = bias[n];
#pragma unroll
      for (int r = 0; r < 8; ++r) {
        const int m = mbase + fm * 16 + r + 8 * hi;   // C layout row
        const float val = acc[fm][fn][r] + bv;
        if (BF16_OUT)
          ((unsigned short*)Cout)[(size_t)m * N + n] = f32_to_bf16_rne(val);
        else
          ((float*)Cout)[(size_t)m * N + n] = val;
      }
    }
}

// ---------------------------------------------------------------------------
// Flash attention over bf16 qkv[T,3072] (layout t x {q,k,v} x h x d).
// Grid: x = seq*16 + qtile(64 rows), y = head. Block 128 thr = 4 waves,
// each wave owns 16 query rows; key loop in tiles of 32.
// K tile: async global->LDS copy (ASYNCcnt). V tile: manual transposed store
// [d][key] so PV's B operand is key-contiguous. P: per-wave C->A LDS bounce.
__global__ __launch_bounds__(128) void attn_kernel(
    const unsigned short* __restrict__ qkv, const int* __restrict__ cu,
    unsigned short* __restrict__ ctx) {
  __shared__ unsigned short lds_k[32 * 64];      // [key][d]
  __shared__ unsigned short lds_v[64 * 32];      // [d][key] (transposed)
  __shared__ unsigned short lds_p[4][16 * 32];   // per-wave P (C->A transpose)

  const int b     = blockIdx.x >> 4;
  const int qtile = blockIdx.x & 15;
  const int h     = blockIdx.y;
  const int t0    = cu[b];
  const int L     = cu[b + 1] - t0;
  const int qbase = qtile * 64;
  if (qbase >= L) return;

  const int lane = threadIdx.x & 31;
  const int wid  = threadIdx.x >> 5;
  const int hi   = lane >> 4;
  const int ln   = lane & 15;

  // Q A-fragments for this wave's 16 rows (clamped for tail rows).
  const int qrow_idx = qbase + wid * 16 + ln;
  const int tq = t0 + imin(qrow_idx, L - 1);
  const unsigned short* qrow = qkv + (size_t)tq * 3072 + h * 64;
  const v16bf qa0 = load_a_frag(qrow, 0, hi);
  const v16bf qa1 = load_a_frag(qrow, 32, hi);

  float m_run[8], l_run[8];
  v8f o[4];
#pragma unroll
  for (int r = 0; r < 8; ++r) { m_run[r] = -1e30f; l_run[r] = 0.f; }
#pragma unroll
  for (int fd = 0; fd < 4; ++fd)
#pragma unroll
    for (int r = 0; r < 8; ++r) o[fd][r] = 0.f;

  const float scale = 0.125f;  // 1/sqrt(64)

  for (int kb = 0; kb < L; kb += 32) {
    __syncthreads();  // previous iteration's LDS reads complete
    {   // cooperative K/V stage: thread -> (key, 16-wide d chunk)
      const int key = threadIdx.x >> 2;
      const int dc  = (threadIdx.x & 3) * 16;
      const int tk  = t0 + imin(kb + key, L - 1);
      // K: async global->LDS (straight copy keeps feature-contiguity for QK^T B)
      const unsigned short* kptr = qkv + (size_t)tk * 3072 + 1024 + h * 64 + dc;
      const unsigned koff =
          (unsigned)(unsigned long long)(&lds_k[key * 64 + dc]);  // LDS offset
      async_copy_b128(koff, kptr);
      async_copy_b128(koff + 16, kptr + 8);
      // V: transposed store so PV's B operand is key-contiguous
      const unsigned short* vptr = qkv + (size_t)tk * 3072 + 2048 + h * 64 + dc;
      const v8u v0 = *(const v8u*)vptr;
      const v8u v1 = *(const v8u*)(vptr + 8);
#pragma unroll
      for (int i = 0; i < 8; ++i) {
        lds_v[(dc + i) * 32 + key]     = v0[i];
        lds_v[(dc + 8 + i) * 32 + key] = v1[i];
      }
    }
    wait_asynccnt0();  // this wave's async K writes landed in LDS
    __syncthreads();   // all waves' K/V tiles visible

    // S = Q K^T for two 16-key subtiles (K B-frags from LDS)
    v8f s[2];
#pragma unroll
    for (int sub = 0; sub < 2; ++sub) {
      const int key = kb + sub * 16 + ln;          // uniform N per lane
      const unsigned short* krow = &lds_k[(sub * 16 + ln) * 64];
      const v16bf kb0 = load_b_frag(krow + hi * 16);
      const v16bf kb1 = load_b_frag(krow + 32 + hi * 16);
      v8f z;
#pragma unroll
      for (int r = 0; r < 8; ++r) z[r] = 0.f;
      z = wmma_bf16(qa0, kb0, z);
      z = wmma_bf16(qa1, kb1, z);
      const float madd = (key < L) ? 0.f : -1e9f;
#pragma unroll
      for (int r = 0; r < 8; ++r) s[sub][r] = z[r] * scale + madd;
    }

    // Online softmax: row r+8*hi lives in a 16-lane group -> shfl width 16.
    float alpha[8];
#pragma unroll
    for (int r = 0; r < 8; ++r) {
      float mx = fmaxf(s[0][r], s[1][r]);
#pragma unroll
      for (int off = 8; off >= 1; off >>= 1)
        mx = fmaxf(mx, __shfl_xor(mx, off, 16));
      const float mnew = fmaxf(m_run[r], mx);
      alpha[r] = __expf(m_run[r] - mnew);
      m_run[r] = mnew;
      const float p0 = __expf(s[0][r] - mnew);
      const float p1 = __expf(s[1][r] - mnew);
      s[0][r] = p0; s[1][r] = p1;
      float ps = p0 + p1;
#pragma unroll
      for (int off = 8; off >= 1; off >>= 1) ps += __shfl_xor(ps, off, 16);
      l_run[r] = l_run[r] * alpha[r] + ps;
    }
#pragma unroll
    for (int fd = 0; fd < 4; ++fd)
#pragma unroll
      for (int r = 0; r < 8; ++r) o[fd][r] *= alpha[r];

    // P (C layout) -> per-wave LDS bf16 [m][key]
#pragma unroll
    for (int r = 0; r < 8; ++r) {
      const int m = r + 8 * hi;
      lds_p[wid][m * 32 + ln]      = f32_to_bf16_rne(s[0][r]);
      lds_p[wid][m * 32 + 16 + ln] = f32_to_bf16_rne(s[1][r]);
    }
    __syncthreads();  // make P visible (uniform trip count per block)

    // P as A operand (16x32), V B-frags from transposed LDS; O += P V
    const unsigned short* prow = &lds_p[wid][ln * 32];
    const v8u plo = *(const v8u*)(prow + hi * 16);
    const v8u phi = *(const v8u*)(prow + hi * 16 + 8);
    v16u pu;
#pragma unroll
    for (int i = 0; i < 8; ++i) { pu[i] = plo[i]; pu[i + 8] = phi[i]; }
    const v16bf pfrag = __builtin_bit_cast(v16bf, pu);
#pragma unroll
    for (int fd = 0; fd < 4; ++fd) {
      const v16bf vb = load_b_frag(&lds_v[(fd * 16 + ln) * 32 + hi * 16]);
      o[fd] = wmma_bf16(pfrag, vb, o[fd]);
    }
  }

  // Epilogue: normalize and write ctx bf16
#pragma unroll
  for (int fd = 0; fd < 4; ++fd) {
#pragma unroll
    for (int r = 0; r < 8; ++r) {
      const int qq = qbase + wid * 16 + r + 8 * hi;
      if (qq < L) {
        const float val = o[fd][r] / l_run[r];
        ctx[(size_t)(t0 + qq) * 1024 + h * 64 + fd * 16 + ln] =
            f32_to_bf16_rne(val);
      }
    }
  }
}

// ---------------------------------------------------------------------------
extern "C" void kernel_launch(void* const* d_in, const int* in_sizes, int n_in,
                              void* d_out, int out_size, void* d_ws, size_t ws_size,
                              hipStream_t stream) {
  const float* hs = (const float*)d_in[0];   // [T,1024] f32
  const float* Wq = (const float*)d_in[1];   // [3072,1024]
  const float* bq = (const float*)d_in[2];   // [3072]
  const float* Wo = (const float*)d_in[3];   // [1024,1024]
  const float* bo = (const float*)d_in[4];   // [1024]
  const int*   cu = (const int*)d_in[5];     // [B+1]
  const int T = in_sizes[0] / 1024;          // 6144 (multiple of 128)
  const int B = in_sizes[5] - 1;             // 8

  // Workspace layout (bf16 staging buffers), ~71 MB total.
  unsigned short* Xb   = (unsigned short*)d_ws;
  unsigned short* Wqb  = Xb  + (size_t)T * 1024;
  unsigned short* Wob  = Wqb + (size_t)3072 * 1024;
  unsigned short* qkvb = Wob + (size_t)1024 * 1024;      // [T,3072]
  unsigned short* ctxb = qkvb + (size_t)T * 3072;        // [T,1024]

  cvt_f32_bf16<<<dim3((T * 1024 + 255) / 256), 256, 0, stream>>>(hs, Xb, T * 1024);
  cvt_f32_bf16<<<dim3((3072 * 1024 + 255) / 256), 256, 0, stream>>>(Wq, Wqb, 3072 * 1024);
  cvt_f32_bf16<<<dim3((1024 * 1024 + 255) / 256), 256, 0, stream>>>(Wo, Wob, 1024 * 1024);

  // QKV projection: [T,3072] bf16  (block tile 128x256)
  gemm_xwT<true><<<dim3(T / 128, 3072 / 256), 256, 0, stream>>>(
      Xb, Wqb, bq, (void*)qkvb, T, 3072, 1024);

  // Attention: grid (B * 16 q-chunks, 16 heads)
  attn_kernel<<<dim3(B * 16, 16), 128, 0, stream>>>(qkvb, cu, ctxb);

  // Output projection: f32 straight into d_out
  gemm_xwT<false><<<dim3(T / 128, 1024 / 256), 256, 0, stream>>>(
      ctxb, Wob, bo, d_out, T, 1024, 1024);
}
